// Optimized_DSTGAT_Attn_84061099917964
// MI455X (gfx1250) — matrix operand order repaired and checked by reading, not compiled, not run on hardware
//
#include <hip/hip_runtime.h>
#include <math.h>

typedef __attribute__((ext_vector_type(16))) _Float16 v16h;
typedef __attribute__((ext_vector_type(8)))  float    v8f;

#define B_   64
#define C_   64
#define T_   1024
#define N_   (B_ * C_)
#define TF_  16
#define FD_  16
#define GC_  32
#define H_   4
#define E_   512
#define ETOT_ (B_ * E_ + N_)   // edges + self loops = 36864

// ---- helpers -------------------------------------------------------------

// A-matrix (16x32 f16) lane layout: element e of v16h in lane L holds
// row M = L%16, K = 16*(L/16) + e   (contiguous, matches B operand fills)
__device__ __forceinline__ int kmapA(int lane, int e) {
  int grp = (lane >> 4) & 1;
  int r = e >> 1;
  (void)r;
  return (grp << 4) + e;
}
__device__ __forceinline__ float eluf(float v)  { return v > 0.f ? v : __expf(v) - 1.f; }
__device__ __forceinline__ float geluf(float x) { return 0.5f * x * (1.f + erff(x * 0.70710678118f)); }
// order-preserving float<->uint for atomicMax-based segment max
__device__ __forceinline__ unsigned fkey(float f) {
  unsigned u = __float_as_uint(f);
  return (u & 0x80000000u) ? ~u : (u | 0x80000000u);
}
__device__ __forceinline__ float funkey(unsigned k) {
  unsigned u = (k & 0x80000000u) ? (k & 0x7FFFFFFFu) : ~k;
  return __uint_as_float(u);
}

// ---- Kernel 1: fused EEGNet temporal stack, one node per block -----------
// conv1(k=25) -> bn+elu -> dw(k=3) -> bn+elu -> sepdw(k=3) -> pw(16x16)
// -> bn+elu -> mean over T.  WMMA for conv1 (im2col) and pointwise.
__global__ __launch_bounds__(128) void feat_kernel(
    const float* __restrict__ x,  const float* __restrict__ w1,
    const float* __restrict__ bn1g, const float* __restrict__ bn1b,
    const float* __restrict__ dw,   const float* __restrict__ bn2g, const float* __restrict__ bn2b,
    const float* __restrict__ sdw,  const float* __restrict__ pw,
    const float* __restrict__ bn3g, const float* __restrict__ bn3b,
    float* __restrict__ feat)
{
  __shared__ float    xs[1072];          // x padded: xs[i] = x[i-14]
  __shared__ _Float16 s1h[16 * 516];     // stage-1 acts (chunk + halo), later s3
  __shared__ _Float16 s2h[16 * 514];     // stage-2 acts (chunk + halo)
  __shared__ float    featL[16];
  __shared__ float b1s[16], b1b[16], b2s[16], b2b[16], b3s[16], b3b[16];

  const int node = blockIdx.x;
  const int tid  = threadIdx.x;
  const int lane = tid & 31, wave = tid >> 5;
  const int j = lane & 15, grp = lane >> 4;
  const float RS = rsqrtf(1.f + 1e-5f);

  if (tid < 16) {
    b1s[tid] = bn1g[tid] * RS; b1b[tid] = bn1b[tid];
    b2s[tid] = bn2g[tid] * RS; b2b[tid] = bn2b[tid];
    b3s[tid] = bn3g[tid] * RS; b3b[tid] = bn3b[tid];
    featL[tid] = 0.f;
  }
  for (int i = tid; i < 1072; i += 128) {
    int t = i - 14;
    xs[i] = (t >= 0 && t < T_) ? x[node * T_ + t] : 0.f;
  }

  // preload WMMA A operands (filter banks), padded K with zeros
  v16h aW, aPW;
  #pragma unroll
  for (int e = 0; e < 16; ++e) {
    int K = kmapA(lane, e);
    aW[e]  = (K < 25) ? (_Float16)w1[j * 25 + K] : (_Float16)0.f;
    aPW[e] = (K < 16) ? (_Float16)pw[j * 16 + K] : (_Float16)0.f;
  }
  float acc8[8];
  #pragma unroll
  for (int r = 0; r < 8; ++r) acc8[r] = 0.f;
  __syncthreads();

  for (int chunk = 0; chunk < 2; ++chunk) {
    const int c0 = chunk * 512;

    // conv1 via WMMA over t in [c0-2, c0+514): tiles of 16 time steps
    for (int i = wave; i < 33; i += 4) {
      int t0 = c0 - 2 + i * 16;
      v16h bx;
      #pragma unroll
      for (int e = 0; e < 16; ++e) {
        int K = (grp << 4) + e;                 // tap index (B row)
        bx[e] = (_Float16)xs[t0 + j + K + 2];   // x[t0+j+K-12] shifted by 14
      }
      v8f c = {};
      v8f d = __builtin_amdgcn_wmma_f32_16x16x32_f16(false, aW, false, bx,
                                                     (short)0, c, false, false);
      int rel = i * 16 + j;                     // offset from c0-2
      if (rel < 516) {
        #pragma unroll
        for (int r = 0; r < 8; ++r) {
          int tf = r + (grp << 3);
          float v = d[r] * b1s[tf] + b1b[tf];
          s1h[tf * 516 + rel] = (_Float16)eluf(v);
        }
      }
    }
    __syncthreads();

    // depthwise conv (k=3) + bn2 + elu -> s2 over [c0-1, c0+513)
    for (int i = tid; i < 16 * 514; i += 128) {
      int ch = i / 514, r2 = i - ch * 514;
      int t = c0 - 1 + r2;
      float a = 0.f;
      #pragma unroll
      for (int k = 0; k < 3; ++k) {
        int tt = t - 1 + k;
        if (tt >= 0 && tt < T_) a += dw[ch * 3 + k] * (float)s1h[ch * 516 + r2 + k];
      }
      s2h[ch * 514 + r2] = (_Float16)eluf(a * b2s[ch] + b2b[ch]);
    }
    __syncthreads();

    // separable depthwise (k=3, raw) -> s3 (reuses s1h) over [c0, c0+512)
    for (int i = tid; i < 16 * 512; i += 128) {
      int ch = i >> 9, r3 = i & 511;
      int t = c0 + r3;
      float a = 0.f;
      #pragma unroll
      for (int k = 0; k < 3; ++k) {
        int tt = t - 1 + k;
        if (tt >= 0 && tt < T_) a += sdw[ch * 3 + k] * (float)s2h[ch * 514 + r3 + k];
      }
      s1h[ch * 516 + r3] = (_Float16)a;
    }
    __syncthreads();

    // pointwise 16x16 matmul via WMMA + bn3 + elu + temporal accumulation
    for (int i = wave; i < 32; i += 4) {
      int t0r = i * 16;
      v16h bx;
      #pragma unroll
      for (int e = 0; e < 16; ++e) {
        int K = (grp << 4) + e;                 // input channel (B row)
        bx[e] = (K < 16) ? s1h[K * 516 + t0r + j] : (_Float16)0.f;
      }
      v8f c = {};
      v8f d = __builtin_amdgcn_wmma_f32_16x16x32_f16(false, aPW, false, bx,
                                                     (short)0, c, false, false);
      #pragma unroll
      for (int r = 0; r < 8; ++r) {
        int fd = r + (grp << 3);
        acc8[r] += eluf(d[r] * b3s[fd] + b3b[fd]);
      }
    }
    __syncthreads();
  }

  #pragma unroll
  for (int r = 0; r < 8; ++r) atomicAdd(&featL[r + (grp << 3)], acc8[r]);
  __syncthreads();
  if (tid < 16) feat[node * 16 + tid] = featL[tid] * (1.f / 1024.f);
}

// ---- Kernel 2: generic WMMA GEMM, M=4096, one 16-row tile per wave -------
// out[M x Ncols] = A[M x Kdim] * B ; B row-major [Kdim x ldb] or, if btrans,
// B[n][k] laid out as Bm[n*ldb + k].
__global__ __launch_bounds__(128) void gemm16_kernel(
    const float* __restrict__ A, int Kdim,
    const float* __restrict__ Bm, int Ncols, int ldb, int btrans,
    float* __restrict__ out, int ldo)
{
  const int lane = threadIdx.x & 31, wave = threadIdx.x >> 5;
  const int rowTile = blockIdx.x * 4 + wave;
  const int j = lane & 15, grp = lane >> 4;
  const int row = rowTile * 16 + j;

  v16h a;
  #pragma unroll
  for (int e = 0; e < 16; ++e) {
    int K = kmapA(lane, e);
    a[e] = (K < Kdim) ? (_Float16)A[row * Kdim + K] : (_Float16)0.f;
  }
  for (int ct = 0; ct < (Ncols >> 4); ++ct) {
    v16h b;
    #pragma unroll
    for (int e = 0; e < 16; ++e) {
      int K = (grp << 4) + e;
      float v = 0.f;
      if (K < Kdim) v = btrans ? Bm[(ct * 16 + j) * ldb + K]
                               : Bm[K * ldb + ct * 16 + j];
      b[e] = (_Float16)v;
    }
    v8f c = {};
    v8f d = __builtin_amdgcn_wmma_f32_16x16x32_f16(false, a, false, b,
                                                   (short)0, c, false, false);
    #pragma unroll
    for (int r = 0; r < 8; ++r)
      out[(rowTile * 16 + r + (grp << 3)) * ldo + ct * 16 + j] = d[r];
  }
}

// ---- GAT edge-softmax pipeline ------------------------------------------
__global__ void attn_prep_kernel(const float* __restrict__ h1,
    const float* __restrict__ as, const float* __restrict__ ad,
    float* __restrict__ asrc, float* __restrict__ adst)
{
  int idx = blockIdx.x * blockDim.x + threadIdx.x;
  if (idx >= N_ * H_) return;
  int n = idx >> 2, h = idx & 3;
  float s1 = 0.f, s2 = 0.f;
  #pragma unroll
  for (int f = 0; f < GC_; ++f) {
    float v = h1[n * 128 + h * 32 + f];
    s1 += v * as[h * 32 + f];
    s2 += v * ad[h * 32 + f];
  }
  asrc[idx] = s1; adst[idx] = s2;
}

__global__ void init_kernel(float* numer, float* z, unsigned* mkey) {
  int idx = blockIdx.x * blockDim.x + threadIdx.x;
  if (idx < N_ * 128) numer[idx] = 0.f;
  if (idx < N_ * H_) { z[idx] = 0.f; mkey[idx] = 0u; }
}

__device__ __forceinline__ void edge_sd(int e, const int* ei, int& s, int& d) {
  if (e < B_ * E_) {
    int b = e >> 9, jj = e & 511;
    s = ei[jj] + b * C_;
    d = ei[E_ + jj] + b * C_;
  } else {
    s = d = e - B_ * E_;       // self loops
  }
}

__global__ void edge_max_kernel(const int* __restrict__ ei,
    const float* __restrict__ asrc, const float* __restrict__ adst,
    unsigned* __restrict__ mkey)
{
  int idx = blockIdx.x * blockDim.x + threadIdx.x;
  if (idx >= ETOT_ * H_) return;
  int e = idx >> 2, h = idx & 3, s, d;
  edge_sd(e, ei, s, d);
  float ev = asrc[s * H_ + h] + adst[d * H_ + h];
  ev = ev > 0.f ? ev : 0.2f * ev;
  atomicMax(&mkey[d * H_ + h], fkey(ev));
}

__global__ void edge_sum_kernel(const int* __restrict__ ei,
    const float* __restrict__ asrc, const float* __restrict__ adst,
    const unsigned* __restrict__ mkey, float* __restrict__ z,
    float* __restrict__ numer, const float* __restrict__ h1)
{
  int idx = blockIdx.x * blockDim.x + threadIdx.x;
  if (idx >= ETOT_ * H_) return;
  int e = idx >> 2, h = idx & 3, s, d;
  edge_sd(e, ei, s, d);
  float ev = asrc[s * H_ + h] + adst[d * H_ + h];
  ev = ev > 0.f ? ev : 0.2f * ev;
  float ex = __expf(ev - funkey(mkey[d * H_ + h]));
  atomicAdd(&z[d * H_ + h], ex);
  int bs = s * 128 + h * 32, bd = d * 128 + h * 32;
  #pragma unroll
  for (int f = 0; f < GC_; ++f) atomicAdd(&numer[bd + f], ex * h1[bs + f]);
}

__global__ void finalize_kernel(const float* __restrict__ numer,
    const float* __restrict__ z, const float* __restrict__ gbias,
    const float* __restrict__ bng, const float* __restrict__ bnb,
    const float* __restrict__ skip, const float* __restrict__ skipb,
    float* __restrict__ out)
{
  int idx = blockIdx.x * blockDim.x + threadIdx.x;
  if (idx >= N_ * GC_) return;
  int n = idx >> 5, g = idx & 31;
  float acc = 0.f;
  #pragma unroll
  for (int h = 0; h < H_; ++h)
    acc += numer[n * 128 + h * 32 + g] / (z[n * 4 + h] + 1e-16f);
  acc = acc * 0.25f + gbias[g];                      // head mean + bias
  acc = acc * (bng[g] * rsqrtf(1.f + 1e-5f)) + bnb[g];  // eval-mode BN
  float sk = skip ? (skip[n * GC_ + g] + (skipb ? skipb[g] : 0.f)) : 0.f;
  out[idx] = geluf(acc + sk);
}

// ---- graph mean pool + classifier ----------------------------------------
__global__ void head_kernel(const float* __restrict__ hf,
    const float* __restrict__ fcw, const float* __restrict__ fcb,
    float* __restrict__ out)
{
  __shared__ float p[GC_];
  int b = blockIdx.x, g = threadIdx.x;
  float s = 0.f;
  for (int c = 0; c < C_; ++c) s += hf[(b * C_ + c) * GC_ + g];
  p[g] = s * (1.f / C_);
  __syncthreads();
  if (g < 2) {
    float o = fcb[g];
    #pragma unroll
    for (int k = 0; k < GC_; ++k) o += p[k] * fcw[g * GC_ + k];
    out[b * 2 + g] = o;
  }
}

// ---- host launcher -------------------------------------------------------
extern "C" void kernel_launch(void* const* d_in, const int* in_sizes, int n_in,
                              void* d_out, int out_size, void* d_ws, size_t ws_size,
                              hipStream_t stream) {
  (void)in_sizes; (void)n_in; (void)out_size; (void)ws_size;
  const float* x        = (const float*)d_in[0];
  const int*   ei       = (const int*)  d_in[1];
  const float* conv1_w  = (const float*)d_in[2];
  const float* bn1_g    = (const float*)d_in[3];
  const float* bn1_b    = (const float*)d_in[4];
  const float* dw_w     = (const float*)d_in[5];
  const float* bn2_g    = (const float*)d_in[6];
  const float* bn2_b    = (const float*)d_in[7];
  const float* sep_dw_w = (const float*)d_in[8];
  const float* sep_pw_w = (const float*)d_in[9];
  const float* bn3_g    = (const float*)d_in[10];
  const float* bn3_b    = (const float*)d_in[11];
  const float* gat1_w   = (const float*)d_in[12];
  const float* gat1_as  = (const float*)d_in[13];
  const float* gat1_ad  = (const float*)d_in[14];
  const float* gat1_bi  = (const float*)d_in[15];
  const float* gat1bn_g = (const float*)d_in[16];
  const float* gat1bn_b = (const float*)d_in[17];
  const float* skip1_w  = (const float*)d_in[18];
  const float* skip1_b  = (const float*)d_in[19];
  const float* gat2_w   = (const float*)d_in[20];
  const float* gat2_as  = (const float*)d_in[21];
  const float* gat2_ad  = (const float*)d_in[22];
  const float* gat2_bi  = (const float*)d_in[23];
  const float* gat2bn_g = (const float*)d_in[24];
  const float* gat2bn_b = (const float*)d_in[25];
  const float* fc_w     = (const float*)d_in[26];
  const float* fc_b     = (const float*)d_in[27];
  float* outp = (float*)d_out;

  float* ws    = (float*)d_ws;
  float* feat  = ws;                    // N*16
  float* hbuf  = feat  + N_ * 16;       // N*128 (reused for both GAT blocks)
  float* skip  = hbuf  + N_ * 128;      // N*32
  float* asrc  = skip  + N_ * GC_;      // N*4
  float* adst  = asrc  + N_ * H_;       // N*4
  unsigned* mkey = (unsigned*)(adst + N_ * H_);   // N*4
  float* z     = (float*)(mkey + N_ * H_);        // N*4
  float* numer = z     + N_ * H_;       // N*128
  float* hf1   = numer + N_ * 128;      // N*32
  float* hf2   = hf1   + N_ * GC_;      // N*32

  // Stage 1: fused temporal feature extractor -> feat[N,16]
  feat_kernel<<<N_, 128, 0, stream>>>(x, conv1_w, bn1_g, bn1_b, dw_w, bn2_g,
                                      bn2_b, sep_dw_w, sep_pw_w, bn3_g, bn3_b, feat);

  // GAT block 1
  gemm16_kernel<<<64, 128, 0, stream>>>(feat, 16, gat1_w, 128, 128, 0, hbuf, 128);
  gemm16_kernel<<<64, 128, 0, stream>>>(feat, 16, skip1_w, 32, 16, 1, skip, 32);
  attn_prep_kernel<<<64, 256, 0, stream>>>(hbuf, gat1_as, gat1_ad, asrc, adst);
  init_kernel<<<2048, 256, 0, stream>>>(numer, z, mkey);
  edge_max_kernel<<<576, 256, 0, stream>>>(ei, asrc, adst, mkey);
  edge_sum_kernel<<<576, 256, 0, stream>>>(ei, asrc, adst, mkey, z, numer, hbuf);
  finalize_kernel<<<512, 256, 0, stream>>>(numer, z, gat1_bi, gat1bn_g, gat1bn_b,
                                           skip, skip1_b, hf1);

  // GAT block 2 (identity skip)
  gemm16_kernel<<<64, 128, 0, stream>>>(hf1, 32, gat2_w, 128, 128, 0, hbuf, 128);
  attn_prep_kernel<<<64, 256, 0, stream>>>(hbuf, gat2_as, gat2_ad, asrc, adst);
  init_kernel<<<2048, 256, 0, stream>>>(numer, z, mkey);
  edge_max_kernel<<<576, 256, 0, stream>>>(ei, asrc, adst, mkey);
  edge_sum_kernel<<<576, 256, 0, stream>>>(ei, asrc, adst, mkey, z, numer, hbuf);
  finalize_kernel<<<512, 256, 0, stream>>>(numer, z, gat2_bi, gat2bn_g, gat2bn_b,
                                           hf1, nullptr, hf2);

  // pool + classifier
  head_kernel<<<B_, 32, 0, stream>>>(hf2, fc_w, fc_b, outp);
}
